// GATRNNHybridModel_74448963109158
// MI455X (gfx1250) — matrix operand order repaired and checked by reading, not compiled, not run on hardware
//
#include <hip/hip_runtime.h>
#include <hip/hip_bf16.h>

typedef _Float16 half_t;
typedef __attribute__((ext_vector_type(8)))  _Float16 v8h;
typedef __attribute__((ext_vector_type(16))) _Float16 v16h;
typedef __attribute__((ext_vector_type(8)))  float    v8f;

#define NNODES 10000
#define NE0    320000
#define ETOT   330000

__device__ __forceinline__ float sigmoidf_(float x) { return 1.f / (1.f + expf(-x)); }
__device__ __forceinline__ int   enc_f(float f) { int i = __float_as_int(f); return i >= 0 ? i : (i ^ 0x7fffffff); }
__device__ __forceinline__ float dec_f(int i)   { return __int_as_float(i >= 0 ? i : (i ^ 0x7fffffff)); }

// ---------------------------------------------------------------------------
// WMMA GEMM: C[M,Npad] = A[M,Kpad](f16, zero-padded) @ B (f16, stored as
// Bt[Npad,Kpad] = B^T, zero-padded) + bias.  One wave per 16x16 tile.
// No bounds checks in the K loop: padding guarantees validity.
// Fragment layouts per CDNA5 ISA 7.12.2 (wave32):
//   A 16x32: lane half hh holds K = hh*8+0..7 and 16+hh*8+0..7  (2x b128)
//   B 32x16: lane half hh holds K = hh*8+0..7 and 16+hh*8+0..7  (2x b128)
//   C/D: VGPR r -> row r (lanes 0-15) / row 8+r (lanes 16-31)
// ---------------------------------------------------------------------------
__global__ __launch_bounds__(32) void gemm_wmma_kernel(
    const half_t* __restrict__ A, int Kpad,
    const half_t* __restrict__ Bt,
    const float* __restrict__ bias,
    float* __restrict__ C, int ldc, int Nact, int relu)
{
  const int lane = threadIdx.x;
  const int hh   = lane >> 4;
  const int l16  = lane & 15;
  const int tile_n = blockIdx.x * 16;
  const int tile_m = blockIdx.y * 16;

  const half_t* Ap = A  + (size_t)(tile_m + l16) * Kpad + hh * 8;
  const half_t* Bp = Bt + (size_t)(tile_n + l16) * Kpad + hh * 8;

  v8f acc = {0.f, 0.f, 0.f, 0.f, 0.f, 0.f, 0.f, 0.f};
  for (int k0 = 0; k0 < Kpad; k0 += 32) {
    v8h a_lo = *(const v8h*)(Ap + k0);
    v8h a_hi = *(const v8h*)(Ap + k0 + 16);
    v8h b_lo = *(const v8h*)(Bp + k0);
    v8h b_hi = *(const v8h*)(Bp + k0 + 16);
    v16h a = __builtin_shufflevector(a_lo, a_hi, 0,1,2,3,4,5,6,7,8,9,10,11,12,13,14,15);
    v16h b = __builtin_shufflevector(b_lo, b_hi, 0,1,2,3,4,5,6,7,8,9,10,11,12,13,14,15);
    acc = __builtin_amdgcn_wmma_f32_16x16x32_f16(false, a, false, b, (short)0, acc, false, false);
  }

  int col = tile_n + l16;
  if (col < Nact) {
    float bb = bias ? bias[col] : 0.f;
#pragma unroll
    for (int r = 0; r < 8; ++r) {
      int row = tile_m + hh * 8 + r;
      float v = acc[r] + bb;
      if (relu) v = fmaxf(v, 0.f);
      C[(size_t)row * ldc + col] = v;
    }
  }
}

// --------------------------- staging kernels -------------------------------
// Activation staging: Y[m*Kpad+k] = f16(X[m*K+k]), zero-padded in K
__global__ void a_stage_kernel(const float* __restrict__ X, half_t* __restrict__ Y,
                               int K, int Kpad, int total)
{
  int idx = blockIdx.x * blockDim.x + threadIdx.x;
  if (idx >= total) return;
  int m = idx / Kpad, k = idx - m * Kpad;
  Y[idx] = (k < K) ? (half_t)X[(size_t)m * K + k] : (half_t)0.f;
}
// LSTM W_ih [N,K] (gates = x @ W_ih^T): Bt[n*Kpad+k] = W[n*K+k]  (padded copy)
__global__ void wih_stage_kernel(const float* __restrict__ W, half_t* __restrict__ Bt,
                                 int K, int Kpad, int total)
{
  int idx = blockIdx.x * blockDim.x + threadIdx.x;
  if (idx >= total) return;
  int n = idx / Kpad, k = idx - n * Kpad;
  Bt[idx] = (k < K) ? (half_t)W[(size_t)n * K + k] : (half_t)0.f;
}
// Dense W [K,N] (y = x @ W): Bt[n*Kpad+k] = W[k*N+n]  (transpose, padded)
__global__ void w_stage_kernel(const float* __restrict__ W, half_t* __restrict__ Bt,
                               int K, int N, int Kpad, int total)
{
  int idx = blockIdx.x * blockDim.x + threadIdx.x;
  if (idx >= total) return;
  int n = idx / Kpad, k = idx - n * Kpad;
  Bt[idx] = (k < K && n < N) ? (half_t)W[(size_t)k * N + n] : (half_t)0.f;
}
__global__ void add2_kernel(const float* __restrict__ a, const float* __restrict__ b,
                            float* __restrict__ o, int n)
{
  int i = blockIdx.x * blockDim.x + threadIdx.x;
  if (i < n) o[i] = a[i] + b[i];
}
__global__ void fill_f32_kernel(float* p, float v, int n)
{
  int i = blockIdx.x * blockDim.x + threadIdx.x;
  if (i < n) p[i] = v;
}
__global__ void fill_i32_kernel(int* p, int v, int n)
{
  int i = blockIdx.x * blockDim.x + threadIdx.x;
  if (i < n) p[i] = v;
}

// --------------------------- LSTM gating -----------------------------------
__global__ void lstm_gate_kernel(const float* __restrict__ gates, float* __restrict__ out,
                                 int ldo, int coloff, int n)
{
  int idx = blockIdx.x * blockDim.x + threadIdx.x;
  if (idx >= n) return;
  int node = idx >> 6, j = idx & 63;
  const float* g = gates + (size_t)node * 256;
  float i_ = g[j], gg = g[128 + j], o_ = g[192 + j];
  float c = sigmoidf_(i_) * tanhf(gg);
  float h = sigmoidf_(o_) * tanhf(c);
  out[(size_t)node * ldo + coloff + j] = h;
}

// --------------------------- GAT pieces ------------------------------------
__global__ void gat_att_kernel(const float* __restrict__ h, const float* __restrict__ atts,
                               const float* __restrict__ attd, float* __restrict__ a_src,
                               float* __restrict__ a_dst, int H, int C, int n)
{
  int idx = blockIdx.x * blockDim.x + threadIdx.x;
  if (idx >= n) return;
  int node = idx / H, hd = idx - node * H;
  const float* hp = h + ((size_t)node * H + hd) * C;
  const float* s = atts + hd * C;
  const float* d = attd + hd * C;
  float x = 0.f, y = 0.f;
  for (int c = 0; c < C; ++c) { float v = hp[c]; x += v * s[c]; y += v * d[c]; }
  a_src[idx] = x;
  a_dst[idx] = y;
}

__global__ void edge_max_kernel(const int* __restrict__ srcA, const int* __restrict__ dstA,
                                const float* __restrict__ a_src, const float* __restrict__ a_dst,
                                int* __restrict__ m, int H, int tot)
{
  int idx = blockIdx.x * blockDim.x + threadIdx.x;
  if (idx >= tot) return;
  int e = idx / H, hd = idx - e * H;
  int s = (e < NE0) ? srcA[e] : (e - NE0);
  int d = (e < NE0) ? dstA[e] : (e - NE0);
  float v = a_src[s * H + hd] + a_dst[d * H + hd];
  v = v > 0.f ? v : 0.2f * v;                       // leaky_relu(0.2)
  atomicMax(&m[d * H + hd], enc_f(v));
}

__global__ void edge_w_kernel(const int* __restrict__ srcA, const int* __restrict__ dstA,
                              const float* __restrict__ a_src, const float* __restrict__ a_dst,
                              const int* __restrict__ m, float* __restrict__ denom,
                              float* __restrict__ wbuf, int H, int tot)
{
  int idx = blockIdx.x * blockDim.x + threadIdx.x;
  if (idx >= tot) return;
  int e = idx / H, hd = idx - e * H;
  int s = (e < NE0) ? srcA[e] : (e - NE0);
  int d = (e < NE0) ? dstA[e] : (e - NE0);
  float v = a_src[s * H + hd] + a_dst[d * H + hd];
  v = v > 0.f ? v : 0.2f * v;
  float w = expf(v - dec_f(m[d * H + hd]));
  wbuf[idx] = w;
  atomicAdd(&denom[d * H + hd], w);
}

__global__ void edge_scatter_kernel(const int* __restrict__ srcA, const int* __restrict__ dstA,
                                    const float* __restrict__ wbuf, const float* __restrict__ denom,
                                    const float* __restrict__ h, float* __restrict__ out,
                                    int H, int C, int tot)
{
  int idx = blockIdx.x * blockDim.x + threadIdx.x;
  if (idx >= tot) return;
  int e = idx / H, hd = idx - e * H;
  int s = (e < NE0) ? srcA[e] : (e - NE0);
  int d = (e < NE0) ? dstA[e] : (e - NE0);
  float alpha = wbuf[idx] / (denom[d * H + hd] + 1e-16f);
  const float* hp = h + ((size_t)s * H + hd) * C;
  float* op = out + ((size_t)d * H + hd) * C;
  for (int c = 0; c < C; ++c) atomicAdd(&op[c], alpha * hp[c]);
}

// --------------------------- LayerNorm (one wave per row) -------------------
__global__ __launch_bounds__(32) void ln_kernel(const float* __restrict__ X,
                                                const float* __restrict__ addb,
                                                const float* __restrict__ g,
                                                const float* __restrict__ b,
                                                float* __restrict__ Y, int D, int relu)
{
  int row = blockIdx.x;
  int lane = threadIdx.x;
  const float* x = X + (size_t)row * D;
  float* y = Y + (size_t)row * D;
  float s = 0.f;
  for (int j = lane; j < D; j += 32) s += x[j] + (addb ? addb[j] : 0.f);
  for (int off = 16; off > 0; off >>= 1) s += __shfl_xor(s, off, 32);
  float mu = s / D;
  float vs = 0.f;
  for (int j = lane; j < D; j += 32) { float v = x[j] + (addb ? addb[j] : 0.f) - mu; vs += v * v; }
  for (int off = 16; off > 0; off >>= 1) vs += __shfl_xor(vs, off, 32);
  float inv = rsqrtf(vs / D + 1e-5f);
  for (int j = lane; j < D; j += 32) {
    float v = (x[j] + (addb ? addb[j] : 0.f) - mu) * inv * g[j] + b[j];
    if (relu) v = fmaxf(v, 0.f);
    y[j] = v;
  }
}

// z[n,0:128] = rnn[n], z[n,128:192] = g[n]
__global__ void concat_kernel(const float* __restrict__ a, const float* __restrict__ b,
                              float* __restrict__ z, int n)
{
  int idx = blockIdx.x * blockDim.x + threadIdx.x;
  if (idx >= n) return;
  int node = idx / 192, j = idx - node * 192;
  z[idx] = (j < 128) ? a[(size_t)node * 128 + j] : b[(size_t)node * 64 + (j - 128)];
}

// ---------------------------------------------------------------------------
static inline int cdiv(int a, int b) { return (a + b - 1) / b; }

extern "C" void kernel_launch(void* const* d_in, const int* in_sizes, int n_in,
                              void* d_out, int out_size, void* d_ws, size_t ws_size,
                              hipStream_t stream)
{
  (void)n_in; (void)out_size; (void)ws_size;

  // ---- input leaf index tables (detect flatten convention) ----
  int t_temporal, t_graphx, t_edge;
  int lstmW[6], lstmB1[6], lstmB2[6];
  int gatW[3], gatAS[3], gatAD[3], gatBB[3];
  int lnG[5], lnB[5];
  int projW, projB, fW[4], fB[4];

  if (in_sizes[0] == NNODES * 4) {
    // recursive insertion-order flatten (dict order from reference file)
    t_temporal = 0; t_graphx = 1; t_edge = 2;
    for (int i = 0; i < 6; ++i) { int b = 3 + i * 3; lstmW[i] = b; lstmB1[i] = b + 1; lstmB2[i] = b + 2; }
    for (int gi = 0; gi < 3; ++gi) { int b = 21 + gi * 4; gatW[gi] = b; gatAS[gi] = b + 1; gatAD[gi] = b + 2; gatBB[gi] = b + 3; }
    for (int k = 0; k < 5; ++k) { lnG[k] = 33 + 2 * k; lnB[k] = 34 + 2 * k; }
    projW = 43; projB = 44;
    for (int k = 0; k < 4; ++k) { fW[k] = 45 + 2 * k; fB[k] = 46 + 2 * k; }
  } else {
    // jax tree_flatten (sorted dict keys at every level)
    t_edge = 0; t_graphx = 1;
    for (int k = 0; k < 4; ++k) { fW[k] = 2 + 2 * k; fB[k] = 3 + 2 * k; }
    for (int gi = 0; gi < 3; ++gi) { int b = 10 + gi * 4; gatW[gi] = b; gatAD[gi] = b + 1; gatAS[gi] = b + 2; gatBB[gi] = b + 3; }
    for (int k = 0; k < 5; ++k) { lnB[k] = 22 + 2 * k; lnG[k] = 23 + 2 * k; }
    for (int i = 0; i < 6; ++i) { int b = 32 + i * 3; lstmW[i] = b; lstmB1[i] = b + 1; lstmB2[i] = b + 2; }
    projW = 50; projB = 51; t_temporal = 52;
  }
  auto F = [&](int i) { return (const float*)d_in[i]; };

  // ---- carve workspace ----
  char* base = (char*)d_ws;
  size_t off = 0;
  auto carve = [&](size_t bytes) -> void* {
    off = (off + 255) & ~(size_t)255;
    void* p = base + off;
    off += bytes;
    return p;
  };
  float*  actA  = (float*)carve((size_t)NNODES * 512 * 4);
  float*  actB  = (float*)carve((size_t)NNODES * 512 * 4);
  float*  hbuf  = (float*)carve((size_t)NNODES * 512 * 4);   // gates / GAT h / z concat
  float*  rnn   = (float*)carve((size_t)NNODES * 128 * 4);
  float*  asrc  = (float*)carve((size_t)NNODES * 8 * 4);
  float*  adst  = (float*)carve((size_t)NNODES * 8 * 4);
  float*  denom = (float*)carve((size_t)NNODES * 8 * 4);
  int*    mmax  = (int*)carve((size_t)NNODES * 8 * 4);
  float*  wedge = (float*)carve((size_t)ETOT * 8 * 4);
  half_t* aF16  = (half_t*)carve((size_t)NNODES * 512 * 2);  // staged GEMM A input
  half_t* wl[6];
  float*  bl[6];
  int insArr[3]  = {4, 128, 128};
  int kpadArr[3] = {32, 128, 128};
  for (int l = 0; l < 3; ++l)
    for (int d = 0; d < 2; ++d) {
      int i = l * 2 + d;
      wl[i] = (half_t*)carve((size_t)256 * kpadArr[l] * 2);
      bl[i] = (float*)carve(256 * 4);
    }
  half_t* wg1   = (half_t*)carve((size_t)512 * 32 * 2);
  half_t* wg2   = (half_t*)carve((size_t)512 * 512 * 2);
  half_t* wg3   = (half_t*)carve((size_t)64 * 512 * 2);
  half_t* wproj = (half_t*)carve((size_t)64 * 64 * 2);
  half_t* wf1   = (half_t*)carve((size_t)128 * 192 * 2);
  half_t* wf2   = (half_t*)carve((size_t)64 * 128 * 2);
  half_t* wf3   = (half_t*)carve((size_t)32 * 64 * 2);
  half_t* wf4   = (half_t*)carve((size_t)16 * 32 * 2);

  // ---- stage weights as padded/transposed f16 Bt[Npad,Kpad] ----
  for (int l = 0; l < 3; ++l)
    for (int d = 0; d < 2; ++d) {
      int i = l * 2 + d;
      int tot = 256 * kpadArr[l];
      wih_stage_kernel<<<cdiv(tot, 256), 256, 0, stream>>>(F(lstmW[i]), wl[i], insArr[l], kpadArr[l], tot);
      add2_kernel<<<1, 256, 0, stream>>>(F(lstmB1[i]), F(lstmB2[i]), bl[i], 256);
    }
  w_stage_kernel<<<cdiv(512 * 32, 256), 256, 0, stream>>>(F(gatW[0]), wg1, 7, 512, 32, 512 * 32);
  w_stage_kernel<<<cdiv(512 * 512, 256), 256, 0, stream>>>(F(gatW[1]), wg2, 512, 512, 512, 512 * 512);
  w_stage_kernel<<<cdiv(64 * 512, 256), 256, 0, stream>>>(F(gatW[2]), wg3, 512, 64, 512, 64 * 512);
  w_stage_kernel<<<cdiv(64 * 64, 256), 256, 0, stream>>>(F(projW), wproj, 64, 64, 64, 64 * 64);
  w_stage_kernel<<<cdiv(128 * 192, 256), 256, 0, stream>>>(F(fW[0]), wf1, 192, 128, 192, 128 * 192);
  w_stage_kernel<<<cdiv(64 * 128, 256), 256, 0, stream>>>(F(fW[1]), wf2, 128, 64, 128, 64 * 128);
  w_stage_kernel<<<cdiv(32 * 64, 256), 256, 0, stream>>>(F(fW[2]), wf3, 64, 32, 64, 32 * 64);
  w_stage_kernel<<<cdiv(16 * 32, 256), 256, 0, stream>>>(F(fW[3]), wf4, 32, 5, 32, 16 * 32);

  auto stageA = [&](const float* X, int K, int Kpad) {
    int tot = NNODES * Kpad;
    a_stage_kernel<<<cdiv(tot, 256), 256, 0, stream>>>(X, aF16, K, Kpad, tot);
  };
  auto gemm = [&](int Kpad, const half_t* Bt, int Npad, const float* bias,
                  float* C, int ldc, int Nact, int relu) {
    dim3 grid(Npad / 16, NNODES / 16);
    gemm_wmma_kernel<<<grid, dim3(32), 0, stream>>>(aF16, Kpad, Bt, bias, C, ldc, Nact, relu);
  };

  // ---- LSTM (seq_len==1, bidirectional, 3 layers) ----
  {
    const float* cur = F(t_temporal);
    int curK = 4;
    float* nexts[3] = {actA, actB, rnn};
    for (int l = 0; l < 3; ++l) {
      float* nxt = nexts[l];
      stageA(cur, curK, kpadArr[l]);
      for (int d = 0; d < 2; ++d) {
        int i = l * 2 + d;
        gemm(kpadArr[l], wl[i], 256, bl[i], hbuf, 256, 256, 0);   // gates
        lstm_gate_kernel<<<cdiv(NNODES * 64, 256), 256, 0, stream>>>(hbuf, nxt, 128, d * 64, NNODES * 64);
      }
      cur = nxt; curK = 128;
    }
  }

  // ---- GAT layers ----
  const int* eidx = (const int*)d_in[t_edge];
  const int* esrc = eidx;
  const int* edst = eidx + NE0;

  auto gat = [&](const float* X, int K, int Kpad, const half_t* W, int H, int C,
                 const float* attS, const float* attD, float* outb) {
    int Fdim = H * C;
    stageA(X, K, Kpad);
    gemm(Kpad, W, Fdim, nullptr, hbuf, Fdim, Fdim, 0);            // h = x @ W
    int nh = NNODES * H;
    gat_att_kernel<<<cdiv(nh, 256), 256, 0, stream>>>(hbuf, attS, attD, asrc, adst, H, C, nh);
    fill_i32_kernel<<<cdiv(nh, 256), 256, 0, stream>>>(mmax, (int)0x80000000, nh);
    fill_f32_kernel<<<cdiv(nh, 256), 256, 0, stream>>>(denom, 0.f, nh);
    fill_f32_kernel<<<cdiv(NNODES * Fdim, 256), 256, 0, stream>>>(outb, 0.f, NNODES * Fdim);
    int eh = ETOT * H;
    edge_max_kernel<<<cdiv(eh, 256), 256, 0, stream>>>(esrc, edst, asrc, adst, mmax, H, eh);
    edge_w_kernel<<<cdiv(eh, 256), 256, 0, stream>>>(esrc, edst, asrc, adst, mmax, denom, wedge, H, eh);
    edge_scatter_kernel<<<cdiv(eh, 256), 256, 0, stream>>>(esrc, edst, wedge, denom, hbuf, outb, H, C, eh);
  };

  gat(F(t_graphx), 7, 32, wg1, 8, 64, F(gatAS[0]), F(gatAD[0]), actA);
  ln_kernel<<<NNODES, 32, 0, stream>>>(actA, F(gatBB[0]), F(lnG[0]), F(lnB[0]), actB, 512, 1);
  gat(actB, 512, 512, wg2, 4, 128, F(gatAS[1]), F(gatAD[1]), actA);
  ln_kernel<<<NNODES, 32, 0, stream>>>(actA, F(gatBB[1]), F(lnG[1]), F(lnB[1]), actB, 512, 1);
  gat(actB, 512, 512, wg3, 1, 64, F(gatAS[2]), F(gatAD[2]), actA);
  ln_kernel<<<NNODES, 32, 0, stream>>>(actA, F(gatBB[2]), F(lnG[2]), F(lnB[2]), actB, 64, 0);

  // graph_proj: relu(g @ Wp + bp)
  stageA(actB, 64, 64);
  gemm(64, wproj, 64, F(projB), actA, 64, 64, 1);

  // ---- head MLP ----
  concat_kernel<<<cdiv(NNODES * 192, 256), 256, 0, stream>>>(rnn, actA, hbuf, NNODES * 192);
  stageA(hbuf, 192, 192);
  gemm(192, wf1, 128, F(fB[0]), actB, 128, 128, 0);
  ln_kernel<<<NNODES, 32, 0, stream>>>(actB, nullptr, F(lnG[3]), F(lnB[3]), actB, 128, 1);
  stageA(actB, 128, 128);
  gemm(128, wf2, 64, F(fB[1]), actA, 64, 64, 0);
  ln_kernel<<<NNODES, 32, 0, stream>>>(actA, nullptr, F(lnG[4]), F(lnB[4]), actA, 64, 1);
  stageA(actA, 64, 64);
  gemm(64, wf3, 32, F(fB[2]), actB, 32, 32, 1);
  stageA(actB, 32, 32);
  gemm(32, wf4, 16, F(fB[3]), (float*)d_out, 5, 5, 0);
}
